// Seq2SeqModel_45638322487915
// MI455X (gfx1250) — compile-verified, matrix operand order checked by reading, not run on hardware
//
#include <hip/hip_runtime.h>
#include <hip/hip_bf16.h>
#include <math.h>

// ---------------- types ----------------
typedef __bf16 bf16_t;
typedef __attribute__((ext_vector_type(8)))  __bf16 v8bf;
typedef __attribute__((ext_vector_type(16))) __bf16 v16bf;
typedef __attribute__((ext_vector_type(8)))  float  v8f;

// ---------------- problem dims ----------------
// D=918, B=64, S=128.  Padded dims: K multiples of 32, N multiples of 64.
#define Dr    918
#define Br    64
#define Sr    128
#define DP    960     // 918 -> 960 (30*32, 15*64)
#define D2r   1836
#define D2P   1856    // 29*64
#define D4r   3672
#define D4P   3712    // 58*64
#define D9r   8262
#define D9P   8320    // 130*64
#define D11r  10098
#define D11P  10112   // 158*64
#define D5r   4590
#define D5P   4608    // 72*64
#define FEATr 8296    // 9D + NE
#define FEATP 8320
#define NEr   34
#define NAr   36
#define NRr   7

// ---------------- workspace layout (bf16 region, element offsets) ----------------
static constexpr long O_ENC   = 0,               S_ENC  = 8192L*DP;
static constexpr long O_WC1   = O_ENC+S_ENC,     S_W1   = (long)DP*DP;
static constexpr long O_WQ1   = O_WC1+S_W1;
static constexpr long O_WC2   = O_WQ1+S_W1;
static constexpr long O_WQ2   = O_WC2+S_W1;
static constexpr long O_WW    = O_WQ2+S_W1,      S_WW   = (long)DP*D9P;
static constexpr long O_CWIH  = O_WW+S_WW,       S_CWIH = (long)D4P*D11P;
static constexpr long O_CWHH  = O_CWIH+S_CWIH,   S_WHH  = (long)D4P*DP;
static constexpr long O_TFWIH = O_CWHH+S_WHH,    S_TWIH = (long)D4P*D2P;
static constexpr long O_TFWHH = O_TFWIH+S_TWIH;
static constexpr long O_TBWIH = O_TFWHH+S_WHH;
static constexpr long O_TBWHH = O_TBWIH+S_TWIH;
static constexpr long O_EFWIH = O_TBWHH+S_WHH,   S_EWIH = (long)D4P*D4P;
static constexpr long O_EFWHH = O_EFWIH+S_EWIH;
static constexpr long O_EBWIH = O_EFWHH+S_WHH;
static constexpr long O_EBWHH = O_EBWIH+S_EWIH;
static constexpr long O_ETW   = O_EBWHH+S_WHH,   S_ETW  = 64L*D5P;
static constexpr long O_ARGW  = O_ETW+S_ETW,     S_AW   = 64L*FEATP;
static constexpr long O_RELW  = O_ARGW+S_AW;
static constexpr long O_HPREV = O_RELW+S_AW,     S_B_DP = (long)Br*DP;
static constexpr long O_REDP  = O_HPREV+S_B_DP;
static constexpr long O_SCUR  = O_REDP+S_B_DP,   S_SCUR = (long)Br*D11P;
static constexpr long O_TRIGIN= O_SCUR+S_SCUR,   S_TIN  = 8192L*D2P;
static constexpr long O_ENTIN = O_TRIGIN+S_TIN,  S_EIN  = 8192L*D4P;
static constexpr long O_ENTOUT= O_ENTIN+S_EIN,   S_EOUT = 8192L*D2P;
static constexpr long O_HA    = O_ENTOUT+S_EOUT;
static constexpr long O_HB    = O_HA+S_B_DP;
static constexpr long O_ETIN  = O_HB+S_B_DP,     S_ETIN = 64L*D5P;
static constexpr long O_FEAT  = O_ETIN+S_ETIN,   S_FEAT = 64L*FEATP;
static constexpr long BF_TOTAL= O_FEAT+S_FEAT;

// ---------------- workspace layout (f32 region) ----------------
static constexpr long F_UH   = 0,              SF_UH = 8192L*DP;
static constexpr long F_WQ   = F_UH+SF_UH,     SF_WQ = (long)Br*DP;
static constexpr long F_E    = F_WQ+SF_WQ,     SF_E  = 8192;
static constexpr long F_RED  = F_E+SF_E,       SF_RD = (long)Br*DP;
static constexpr long F_GCEL = F_RED+SF_RD,    SF_G  = (long)Br*D4P;
static constexpr long F_XA   = F_GCEL+SF_G,    SF_X  = 8192L*D4P;
static constexpr long F_XB   = F_XA+SF_X;
static constexpr long F_C    = F_XB+SF_X,      SF_C  = (long)Br*Dr;
static constexpr long F_SC   = F_C+SF_C,       SF_SC = 4L*8192;
static constexpr long F_EVT  = F_SC+SF_SC,     SF_S64= 64L*64;
static constexpr long F_ARG  = F_EVT+SF_S64;
static constexpr long F_REL  = F_ARG+SF_S64;

// ---------------- output layout (f32 elements) ----------------
static constexpr long OUT_REL=0, OUT_TSW=448, OUT_TEW=8640, OUT_ESW=16832, OUT_EEW=25024,
  OUT_HID=33216, OUT_CELL=91968, OUT_TET=150720, OUT_EAT=385728, OUT_EVT=620736, OUT_ARG=622912;

// ---------------- helpers ----------------
__device__ inline v8f vzero8f() {
  v8f z;
  #pragma unroll
  for (int i=0;i<8;++i) z[i]=0.f;
  return z;
}

__device__ inline float sigf(float x){ return 1.f/(1.f+__expf(-x)); }

// A/B fragment loader for v_wmma_f32_16x16x32_bf16.
// lane<16: K[k..k+7] and K[k+16..k+23]; lane>=16: K[k+8..k+15] and K[k+24..k+31].
__device__ inline v16bf load_frag(const bf16_t* base, int ld, int row, int k, int lane){
  int r  = row + (lane & 15);
  int kk = k + (lane >> 4) * 8;
  const bf16_t* p = base + (long)r*ld + kk;
  v8bf lo = *(const v8bf*)p;
  v8bf hi = *(const v8bf*)(p + 16);
  v16bf f;
  #pragma unroll
  for (int i=0;i<8;++i){ f[i]=lo[i]; f[i+8]=hi[i]; }
  return f;
}

// ---------------- generic bf16 WMMA GEMM: C[MxNP](f32) = A[MxKP] * B[NPxKP]^T (+bias, +=) ----
// One wave per 32x64 tile. grid.x = NP/64, grid.y = M/32.
__global__ __launch_bounds__(32)
void k_gemm(const bf16_t* __restrict__ A, const bf16_t* __restrict__ B, float* __restrict__ C,
            int M, int N, int KP, int lda, int ldb, int ldc, const float* __restrict__ bias, int accum){
  const int lane = threadIdx.x;
  const int m0 = blockIdx.y*32, n0 = blockIdx.x*64;
  v8f acc[2][4];
  #pragma unroll
  for (int i=0;i<2;++i)
    for (int j=0;j<4;++j) acc[i][j]=vzero8f();
  for (int k=0;k<KP;k+=32){
    v16bf a0 = load_frag(A, lda, m0,    k, lane);
    v16bf a1 = load_frag(A, lda, m0+16, k, lane);
    #pragma unroll
    for (int j=0;j<4;++j){
      v16bf bf = load_frag(B, ldb, n0+j*16, k, lane);
      acc[0][j]=__builtin_amdgcn_wmma_f32_16x16x32_bf16(false,a0,false,bf,(short)0,acc[0][j],false,false);
      acc[1][j]=__builtin_amdgcn_wmma_f32_16x16x32_bf16(false,a1,false,bf,(short)0,acc[1][j],false,false);
    }
  }
  const int half = lane>>4, cl = lane&15;
  #pragma unroll
  for (int i=0;i<2;++i){
    #pragma unroll
    for (int j=0;j<4;++j){
      int n = n0 + j*16 + cl;
      float bv = (bias && n < N) ? bias[n] : 0.f;
      #pragma unroll
      for (int e=0;e<8;++e){
        int m = m0 + i*16 + e + half*8;
        long id = (long)m*ldc + n;
        float v = acc[i][j][e] + bv;
        if (accum) v += C[id];
        C[id] = v;
      }
    }
  }
}

// ---------------- fused recurrent LSTM step: g = h_prev @ Whh^T (4 gates) + xproj; gates ------
// grid = (58, 4), one wave per 16x16 column tile of D, computing all 4 gate tiles.
__global__ __launch_bounds__(32)
void k_lstm_step(const float* __restrict__ xproj, const bf16_t* __restrict__ hprev,
                 const bf16_t* __restrict__ Whh, float* __restrict__ c,
                 bf16_t* __restrict__ hnext, bf16_t* __restrict__ hout,
                 int ldhout, int coloff, int s, int ldx){
  const int lane = threadIdx.x;
  const int n0 = blockIdx.x*16, m0 = blockIdx.y*16;
  v8f acc[4];
  #pragma unroll
  for (int g=0; g<4; ++g) acc[g]=vzero8f();
  for (int k=0;k<DP;k+=32){
    v16bf a = load_frag(hprev, DP, m0, k, lane);
    #pragma unroll
    for (int g=0; g<4; ++g){
      v16bf bf = load_frag(Whh, DP, g*Dr + n0, k, lane);
      acc[g]=__builtin_amdgcn_wmma_f32_16x16x32_bf16(false,a,false,bf,(short)0,acc[g],false,false);
    }
  }
  const int half = lane>>4, col = n0 + (lane&15);
  if (col < Dr){
    #pragma unroll
    for (int e=0;e<8;++e){
      int m = m0 + e + half*8;
      const float* xr = xproj + ((long)m*Sr + s)*ldx;
      float gi = acc[0][e] + xr[col];
      float gf = acc[1][e] + xr[Dr+col];
      float gg = acc[2][e] + xr[2*Dr+col];
      float go = acc[3][e] + xr[3*Dr+col];
      float cn = sigf(gf)*c[m*Dr+col] + sigf(gi)*tanhf(gg);
      c[m*Dr+col] = cn;
      float h = sigf(go)*tanhf(cn);
      hnext[(long)m*DP + col] = (bf16_t)h;
      hout[((long)m*Sr + s)*ldhout + coloff + col] = (bf16_t)h;
    }
  }
}

// ---------------- pack f32 -> padded bf16 ----------------
__global__ void k_cvt(const float* __restrict__ src, bf16_t* __restrict__ dst,
                      int rows, int cols, int sld, int dld, int coff){
  long idx = (long)blockIdx.x*256 + threadIdx.x;
  long tot = (long)rows*cols;
  if (idx >= tot) return;
  int r = (int)(idx / cols), cc = (int)(idx % cols);
  dst[(long)r*dld + coff + cc] = (bf16_t)src[(long)r*sld + cc];
}

// broadcast hidden[b] over S into packed [B*S x dld]
__global__ void k_bcast(const float* __restrict__ src, bf16_t* __restrict__ dst,
                        int cols, int sld, int dld, int coff){
  long idx = (long)blockIdx.x*256 + threadIdx.x;
  long tot = 8192L*cols;
  if (idx >= tot) return;
  int r = (int)(idx / cols), cc = (int)(idx % cols);
  dst[(long)r*dld + coff + cc] = (bf16_t)src[(long)(r>>7)*sld + cc];
}

// ---------------- attention energy: e[b,s] = sum_d tanh(wq[b,d]+uh[b*S+s,d])*v[d] + bv -------
__global__ __launch_bounds__(32)
void k_attn_e(const float* __restrict__ uh, const float* __restrict__ wq,
              const float* __restrict__ v, const float* __restrict__ bv,
              const unsigned char* __restrict__ mask, float* __restrict__ e, int ldu){
  int row = blockIdx.x, lane = threadIdx.x;
  int b = row >> 7;
  const float* uhr = uh + (long)row*ldu;
  const float* wqr = wq + (long)b*ldu;
  float s = 0.f;
  for (int d=lane; d<Dr; d+=32) s += tanhf(wqr[d]+uhr[d])*v[d];
  #pragma unroll
  for (int o=16;o;o>>=1) s += __shfl_xor(s,o,32);
  if (lane==0) e[row] = mask[row] ? -INFINITY : (s + bv[0]);
}

// softmax over S=128 + ctx = a . enc_hs, ctx packed bf16 into scur
__global__ __launch_bounds__(128)
void k_attn_ctx(const float* __restrict__ e, const float* __restrict__ enc,
                bf16_t* __restrict__ scur, int ctxoff, int lds){
  __shared__ float sa[128]; __shared__ float sh[4];
  int b = blockIdx.x, t = threadIdx.x;
  float ev = e[b*Sr + t];
  float m = ev;
  #pragma unroll
  for (int o=16;o;o>>=1) m = fmaxf(m, __shfl_xor(m,o,32));
  if ((t&31)==0) sh[t>>5]=m;
  __syncthreads();
  float mx = fmaxf(fmaxf(sh[0],sh[1]),fmaxf(sh[2],sh[3]));
  __syncthreads();
  float x = __expf(ev-mx), sm = x;
  #pragma unroll
  for (int o=16;o;o>>=1) sm += __shfl_xor(sm,o,32);
  if ((t&31)==0) sh[t>>5]=sm;
  __syncthreads();
  float tot = sh[0]+sh[1]+sh[2]+sh[3];
  __syncthreads();
  sa[t] = x/tot;
  __syncthreads();
  const float* encb = enc + (long)b*Sr*Dr;
  for (int col=t; col<Dr; col+=128){
    float acc = 0.f;
    for (int s2=0;s2<Sr;++s2) acc += sa[s2]*encb[(long)s2*Dr + col];
    scur[(long)b*lds + ctxoff + col] = (bf16_t)acc;
  }
}

// ---------------- LSTMCell gates ----------------
__global__ void k_cell_gates(const float* __restrict__ g, const float* __restrict__ cprev,
                             float* __restrict__ hid, float* __restrict__ cell, int ldg){
  int idx = blockIdx.x*256 + threadIdx.x;
  if (idx >= Br*Dr) return;
  int m = idx/Dr, j = idx%Dr;
  const float* gr = g + (long)m*ldg;
  float c = sigf(gr[Dr+j])*cprev[idx] + sigf(gr[j])*tanhf(gr[2*Dr+j]);
  hid[idx] = sigf(gr[3*Dr+j])*tanhf(c);
  cell[idx] = c;
}

// ---------------- pointer score: sc[row] = out[row,:2D] . W + b (masked) ----------------
__global__ __launch_bounds__(32)
void k_score(const bf16_t* __restrict__ out, int ldo, int coff,
             const float* __restrict__ W, const float* __restrict__ b,
             const unsigned char* __restrict__ mask, float* __restrict__ sc){
  int row = blockIdx.x, lane = threadIdx.x;
  const bf16_t* r = out + (long)row*ldo + coff;
  float s = 0.f;
  for (int j=lane; j<D2r; j+=32) s += (float)r[j]*W[j];
  #pragma unroll
  for (int o=16;o;o>>=1) s += __shfl_xor(s,o,32);
  if (lane==0) sc[row] = mask[row] ? -INFINITY : (s + b[0]);
}

__global__ __launch_bounds__(128)
void k_softmax128(const float* __restrict__ sc, float* __restrict__ out){
  __shared__ float sh[4];
  int b = blockIdx.x, t = threadIdx.x;
  float v = sc[b*Sr + t];
  float m = v;
  #pragma unroll
  for (int o=16;o;o>>=1) m = fmaxf(m, __shfl_xor(m,o,32));
  if ((t&31)==0) sh[t>>5]=m;
  __syncthreads();
  float mx = fmaxf(fmaxf(sh[0],sh[1]),fmaxf(sh[2],sh[3]));
  __syncthreads();
  float x = __expf(v-mx), s = x;
  #pragma unroll
  for (int o=16;o;o>>=1) s += __shfl_xor(s,o,32);
  if ((t&31)==0) sh[t>>5]=s;
  __syncthreads();
  out[b*Sr+t] = x/(sh[0]+sh[1]+sh[2]+sh[3]);
}

// weighted sums: dst[b, j<2D] = sum_s wEnd*out ; dst[b, j>=2D] = sum_s wStart*out
__global__ __launch_bounds__(256)
void k_wsum(const float* __restrict__ wEnd, const float* __restrict__ wStart,
            const bf16_t* __restrict__ out, int ldo, int coff, float* __restrict__ dst,
            bf16_t* p1, int ld1, int o1, bf16_t* p2, int ld2, int o2){
  int b = blockIdx.x;
  const bf16_t* ob = out + (long)b*Sr*ldo + coff;
  for (int j=threadIdx.x; j<D4r; j+=256){
    const float* w = (j < D2r) ? wEnd : wStart;
    int jj = (j < D2r) ? j : j - D2r;
    float acc = 0.f;
    for (int s=0;s<Sr;++s) acc += w[b*Sr+s]*(float)ob[(long)s*ldo + jj];
    dst[(long)b*D4r + j] = acc;
    if (p1) p1[(long)b*ld1 + o1 + j] = (bf16_t)acc;
    if (p2) p2[(long)b*ld2 + o2 + j] = (bf16_t)acc;
  }
}

// log_softmax over n<=64 values; optionally packs RAW logits into feat (bf16)
__global__ __launch_bounds__(64)
void k_logsm(const float* __restrict__ in, int ldin, int n,
             float* __restrict__ out, int ldout, bf16_t* pack, int ldp, int off){
  __shared__ float buf[64]; __shared__ float lse;
  int b = blockIdx.x, t = threadIdx.x;
  float v = (t < n) ? in[(long)b*ldin + t] : -INFINITY;
  buf[t] = v;
  __syncthreads();
  if (t==0){
    float m=-INFINITY; for (int i=0;i<n;++i) m=fmaxf(m,buf[i]);
    float s=0.f;       for (int i=0;i<n;++i) s+=__expf(buf[i]-m);
    lse = m + __logf(s);
  }
  __syncthreads();
  if (t < n){
    out[(long)b*ldout + t] = v - lse;
    if (pack) pack[(long)b*ldp + off + t] = (bf16_t)v;
  }
}

// ---------------- host orchestration ----------------
extern "C" void kernel_launch(void* const* d_in, const int* in_sizes, int n_in,
                              void* d_out, int out_size, void* d_ws, size_t ws_size,
                              hipStream_t stream){
  (void)in_sizes; (void)n_in; (void)out_size; (void)ws_size;
  const float* ptup = (const float*)d_in[1];
  const float* hph  = (const float*)d_in[2];
  const float* hpc  = (const float*)d_in[3];
  const float* enc  = (const float*)d_in[4];
  const unsigned char* mask = (const unsigned char*)d_in[5];
  auto P = [&](int i){ return (const float*)d_in[i]; };

  bf16_t* WB = (bf16_t*)d_ws;
  float*  WF = (float*)((char*)d_ws + BF_TOTAL*sizeof(bf16_t));
  float*  out = (float*)d_out;

  auto CVT = [&](const float* s, bf16_t* d, long rows, long cols, int sld, int dld, int coff){
    long tot = rows*cols;
    k_cvt<<<dim3((unsigned)((tot+255)/256)),256,0,stream>>>(s,d,(int)rows,(int)cols,sld,dld,coff);
  };
  auto GEMM = [&](const bf16_t* A,int lda,const bf16_t* B,int ldb,float* C,int ldc,
                  int M,int N,int KP,const float* bias,int acc){
    k_gemm<<<dim3(ldc/64, M/32),32,0,stream>>>(A,B,C,M,N,KP,lda,ldb,ldc,bias,acc);
  };

  // 0) zero whole bf16 region (establishes all padding) ------------------------------
  (void)hipMemsetAsync(WB, 0, (size_t)BF_TOTAL*sizeof(bf16_t), stream);

  // 1) pack weights + activations ----------------------------------------------------
  CVT(enc,   WB+O_ENC,   8192, Dr,   Dr,   DP,   0);
  CVT(P(10), WB+O_WC1,   Dr,   Dr,   Dr,   DP,   0);
  CVT(P(11), WB+O_WQ1,   Dr,   Dr,   Dr,   DP,   0);
  CVT(P(15), WB+O_WC2,   Dr,   Dr,   Dr,   DP,   0);
  CVT(P(16), WB+O_WQ2,   Dr,   Dr,   Dr,   DP,   0);
  CVT(P(20), WB+O_WW,    Dr,   D9r,  D9r,  D9P,  0);
  CVT(P(22), WB+O_CWIH,  D4r,  D11r, D11r, D11P, 0);
  CVT(P(23), WB+O_CWHH,  D4r,  Dr,   Dr,   DP,   0);
  CVT(P(25), WB+O_TFWIH, D4r,  D2r,  D2r,  D2P,  0);
  CVT(P(26), WB+O_TFWHH, D4r,  Dr,   Dr,   DP,   0);
  CVT(P(28), WB+O_TBWIH, D4r,  D2r,  D2r,  D2P,  0);
  CVT(P(29), WB+O_TBWHH, D4r,  Dr,   Dr,   DP,   0);
  CVT(P(31), WB+O_EFWIH, D4r,  D4r,  D4r,  D4P,  0);
  CVT(P(32), WB+O_EFWHH, D4r,  Dr,   Dr,   DP,   0);
  CVT(P(34), WB+O_EBWIH, D4r,  D4r,  D4r,  D4P,  0);
  CVT(P(35), WB+O_EBWHH, D4r,  Dr,   Dr,   DP,   0);
  CVT(P(45), WB+O_ETW,   NEr,  D5r,  D5r,  D5P,  0);
  CVT(P(47), WB+O_ARGW,  NAr,  FEATr,FEATr,FEATP,0);
  CVT(P(49), WB+O_RELW,  NRr,  FEATr,FEATr,FEATP,0);
  CVT(hph,   WB+O_HPREV, Br,   Dr,   Dr,   DP,   0);
  CVT(ptup,  WB+O_SCUR,  Br,   D9r,  D9r,  D11P, 0);

  // 2) attention 1 (query = h_prev) --------------------------------------------------
  GEMM(WB+O_ENC,DP,   WB+O_WC1,DP, WF+F_UH,DP, 8192,Dr,DP, nullptr,0);
  GEMM(WB+O_HPREV,DP, WB+O_WQ1,DP, WF+F_WQ,DP, Br,  Dr,DP, P(12),0);
  k_attn_e<<<8192,32,0,stream>>>(WF+F_UH, WF+F_WQ, P(13), P(14), mask, WF+F_E, DP);
  k_attn_ctx<<<Br,128,0,stream>>>(WF+F_E, enc, WB+O_SCUR, 9*Dr, D11P);

  // 3) red = prev_tuples @ w_W^T + w_b; pack ----------------------------------------
  GEMM(WB+O_SCUR,D11P, WB+O_WW,D9P, WF+F_RED,DP, Br,Dr,D9P, P(21),0);
  CVT(WF+F_RED, WB+O_REDP, Br, Dr, DP, DP, 0);

  // 4) attention 2 (query = red) -----------------------------------------------------
  GEMM(WB+O_ENC,DP,  WB+O_WC2,DP, WF+F_UH,DP, 8192,Dr,DP, nullptr,0);
  GEMM(WB+O_REDP,DP, WB+O_WQ2,DP, WF+F_WQ,DP, Br,  Dr,DP, P(17),0);
  k_attn_e<<<8192,32,0,stream>>>(WF+F_UH, WF+F_WQ, P(18), P(19), mask, WF+F_E, DP);
  k_attn_ctx<<<Br,128,0,stream>>>(WF+F_E, enc, WB+O_SCUR, 10*Dr, D11P);

  // 5) LSTMCell ----------------------------------------------------------------------
  GEMM(WB+O_SCUR,D11P, WB+O_CWIH,D11P, WF+F_GCEL,D4P, Br,D4r,D11P, P(24),0);
  GEMM(WB+O_HPREV,DP,  WB+O_CWHH,DP,   WF+F_GCEL,D4P, Br,D4r,DP,   nullptr,1);
  k_cell_gates<<<(Br*Dr+255)/256,256,0,stream>>>(WF+F_GCEL, hpc, out+OUT_HID, out+OUT_CELL, D4P);

  // 6) build trig_in [enc|hidden] and ent_in [enc|hidden|trig_out...] ----------------
  CVT(enc, WB+O_TRIGIN, 8192, Dr, Dr, D2P, 0);
  k_bcast<<<(unsigned)((8192L*Dr+255)/256),256,0,stream>>>(out+OUT_HID, WB+O_TRIGIN, Dr, Dr, D2P, Dr);
  CVT(enc, WB+O_ENTIN, 8192, Dr, Dr, D4P, 0);
  k_bcast<<<(unsigned)((8192L*Dr+255)/256),256,0,stream>>>(out+OUT_HID, WB+O_ENTIN, Dr, Dr, D4P, Dr);

  auto SCAN = [&](const float* xp, const bf16_t* whh, bf16_t* hout, int ldho, int coff, bool fwd){
    (void)hipMemsetAsync(WB+O_HA, 0, (size_t)S_B_DP*sizeof(bf16_t), stream);
    (void)hipMemsetAsync(WB+O_HB, 0, (size_t)S_B_DP*sizeof(bf16_t), stream);
    (void)hipMemsetAsync(WF+F_C,  0, (size_t)SF_C*sizeof(float), stream);
    bf16_t* hp = WB+O_HA; bf16_t* hn = WB+O_HB;
    for (int t=0;t<Sr;++t){
      int s = fwd ? t : (Sr-1-t);
      k_lstm_step<<<dim3(58,4),32,0,stream>>>(xp, hp, whh, WF+F_C, hn, hout, ldho, coff, s, D4P);
      bf16_t* tmp = hp; hp = hn; hn = tmp;
    }
  };

  // 7) trig BiLSTM: input proj then scans (writes trig_out into ent_in cols [2D..4D)) -
  GEMM(WB+O_TRIGIN,D2P, WB+O_TFWIH,D2P, WF+F_XA,D4P, 8192,D4r,D2P, P(27),0);
  GEMM(WB+O_TRIGIN,D2P, WB+O_TBWIH,D2P, WF+F_XB,D4P, 8192,D4r,D2P, P(30),0);
  SCAN(WF+F_XA, WB+O_TFWHH, WB+O_ENTIN, D4P, D2r,     true);
  SCAN(WF+F_XB, WB+O_TBWHH, WB+O_ENTIN, D4P, D2r+Dr,  false);

  // 8) ent BiLSTM --------------------------------------------------------------------
  GEMM(WB+O_ENTIN,D4P, WB+O_EFWIH,D4P, WF+F_XA,D4P, 8192,D4r,D4P, P(33),0);
  GEMM(WB+O_ENTIN,D4P, WB+O_EBWIH,D4P, WF+F_XB,D4P, 8192,D4r,D4P, P(36),0);
  SCAN(WF+F_XA, WB+O_EFWHH, WB+O_ENTOUT, D2P, 0,  true);
  SCAN(WF+F_XB, WB+O_EBWHH, WB+O_ENTOUT, D2P, Dr, false);

  // 9) pointer scores + softmaxes ----------------------------------------------------
  k_score<<<8192,32,0,stream>>>(WB+O_ENTIN, D4P, D2r, P(37), P(38), mask, WF+F_SC+0);
  k_score<<<8192,32,0,stream>>>(WB+O_ENTIN, D4P, D2r, P(39), P(40), mask, WF+F_SC+8192);
  k_score<<<8192,32,0,stream>>>(WB+O_ENTOUT,D2P, 0,   P(41), P(42), mask, WF+F_SC+16384);
  k_score<<<8192,32,0,stream>>>(WB+O_ENTOUT,D2P, 0,   P(43), P(44), mask, WF+F_SC+24576);
  k_softmax128<<<Br,128,0,stream>>>(WF+F_SC+0,     out+OUT_TSW);
  k_softmax128<<<Br,128,0,stream>>>(WF+F_SC+8192,  out+OUT_TEW);
  k_softmax128<<<Br,128,0,stream>>>(WF+F_SC+16384, out+OUT_ESW);
  k_softmax128<<<Br,128,0,stream>>>(WF+F_SC+24576, out+OUT_EEW);

  // 10) attention-pooled span vectors ------------------------------------------------
  k_wsum<<<Br,256,0,stream>>>(out+OUT_TEW, out+OUT_TSW, WB+O_ENTIN, D4P, D2r, out+OUT_TET,
                              WB+O_ETIN, D5P, 0, WB+O_FEAT, FEATP, 0);
  k_wsum<<<Br,256,0,stream>>>(out+OUT_EEW, out+OUT_ESW, WB+O_ENTOUT, D2P, 0, out+OUT_EAT,
                              WB+O_FEAT, FEATP, D4r, nullptr, 0, 0);
  CVT(out+OUT_HID, WB+O_ETIN, Br, Dr, Dr, D5P,  D4r);
  CVT(out+OUT_HID, WB+O_FEAT, Br, Dr, Dr, FEATP, D4r+D4r+NEr);

  // 11) classifier heads -------------------------------------------------------------
  GEMM(WB+O_ETIN,D5P, WB+O_ETW,D5P, WF+F_EVT,64, Br,NEr,D5P, P(46),0);
  k_logsm<<<Br,64,0,stream>>>(WF+F_EVT,64,NEr, out+OUT_EVT,NEr, WB+O_FEAT,FEATP,2*D4r);
  GEMM(WB+O_FEAT,FEATP, WB+O_ARGW,FEATP, WF+F_ARG,64, Br,NAr,FEATP, P(48),0);
  GEMM(WB+O_FEAT,FEATP, WB+O_RELW,FEATP, WF+F_REL,64, Br,NRr,FEATP, P(50),0);
  k_logsm<<<Br,64,0,stream>>>(WF+F_ARG,64,NAr, out+OUT_ARG,NAr, nullptr,0,0);
  k_logsm<<<Br,64,0,stream>>>(WF+F_REL,64,NRr, out+OUT_REL,NRr, nullptr,0,0);
}
